// CausalSelfAttention_29781303230752
// MI455X (gfx1250) — matrix operand order repaired and checked by reading, not compiled, not run on hardware
//
#include <hip/hip_runtime.h>
#include <hip/hip_bf16.h>

typedef __attribute__((ext_vector_type(16))) __bf16 v16bf;
typedef __attribute__((ext_vector_type(8)))  float  v8f;
typedef __attribute__((ext_vector_type(4)))  unsigned int u32x4;
typedef __attribute__((ext_vector_type(8)))  unsigned int u32x8;

// ---------------------------------------------------------------------------
// WMMA fragment helpers (CDNA5 16x16x32 bf16, wave32), all 16B-aligned
// vector accesses so LDS reads lower to ds_load_b128.
// A fragment (16x32 row-major, ld elements): lane L: row=L&15, kb=(L<16?0:8),
//   elements 0..7 -> K=kb+i, 8..15 -> K=kb+16+(i-8): two contiguous 8-element
//   runs at p and p+16.
// B fragment from bt[n][k] storage: lane L: col=L&15, kb=(L<16?0:8),
//   same K order as A: two 8-element runs at bt + col*ld + kb and +16.
// C/D layout: VGPR r -> M = r + 8*(lane>=16), N = lane&15.
// ---------------------------------------------------------------------------
union frag_cast { uint4 u[2]; v16bf v; };

__device__ inline v16bf load_a_frag(const __bf16* __restrict__ base, int ld) {
    int lane = threadIdx.x & 31;
    const __bf16* p = base + (lane & 15) * ld + ((lane < 16) ? 0 : 8);
    frag_cast f;
    f.u[0] = *(const uint4*)(p);        // K = kb + 0..7
    f.u[1] = *(const uint4*)(p + 16);   // K = kb + 16..23
    return f.v;
}

__device__ inline v16bf load_bt_frag(const __bf16* __restrict__ bt, int ld) {
    int lane = threadIdx.x & 31;
    const __bf16* p = bt + (lane & 15) * ld + ((lane < 16) ? 0 : 8);
    frag_cast f;
    f.u[0] = *(const uint4*)(p);
    f.u[1] = *(const uint4*)(p + 16);
    return f.v;
}

__device__ inline v8f wmma_bf16(v16bf a, v16bf b, v8f c) {
    return __builtin_amdgcn_wmma_f32_16x16x32_bf16(false, a, false, b,
                                                   (short)0, c, false, false);
}

__device__ inline unsigned lds_offset(const void* p) {
    // LDS flat addresses carry the byte offset in addr[31:0]
    return (unsigned)(unsigned long long)p;
}

// ---------------------------------------------------------------------------
// f32 -> bf16 conversion, 8 elements / thread (all sizes divisible by 8)
// ---------------------------------------------------------------------------
__global__ void f32_to_bf16_kernel(const float* __restrict__ in,
                                   __bf16* __restrict__ out, int n8) {
    int i = blockIdx.x * blockDim.x + threadIdx.x;
    if (i >= n8) return;
    const float4* src = (const float4*)in + i * 2;
    float4 a = src[0], b = src[1];
    union { uint4 u; __bf16 h[8]; } pk;
    pk.h[0] = (__bf16)a.x; pk.h[1] = (__bf16)a.y;
    pk.h[2] = (__bf16)a.z; pk.h[3] = (__bf16)a.w;
    pk.h[4] = (__bf16)b.x; pk.h[5] = (__bf16)b.y;
    pk.h[6] = (__bf16)b.z; pk.h[7] = (__bf16)b.w;
    ((uint4*)out)[i] = pk.u;
}

// ---------------------------------------------------------------------------
// Tiled bf16 GEMM: C[M,N] = A[M,K] * B[K,N].  128 threads = 4 waves.
// Tile 128x64, BK=32.  Wave w owns rows w*32..w*32+31 (two 16-row sub-tiles),
// 8 WMMAs per k-step.  A tile (8KB, row-major) fetched by the Tensor Data
// Mover (tensor_load_to_lds, TENSORcnt); B tile via uint4 global loads with
// incremental pointers + transposed ds_store_b16.
// mode 0: f32 C.   mode 1: fused RoPE epilogue (N-tile == head dim, rotate
// partner d^32 is acc[sub][nt^2] in the same lane).
// ---------------------------------------------------------------------------
__global__ __launch_bounds__(128) void gemm_bf16_kernel(
    const __bf16* __restrict__ A, const __bf16* __restrict__ B,
    int M, int N, int K, int mode,
    float* __restrict__ C,
    __bf16* __restrict__ Qb, __bf16* __restrict__ Kb, __bf16* __restrict__ Vb,
    const float* __restrict__ cosp, const float* __restrict__ sinp)
{
    __shared__ __bf16 As[128 * 32];   // [row][k]
    __shared__ __bf16 Bt[64 * 32];    // [n][k] (transposed)

    const int tid  = threadIdx.x;
    const int lane = tid & 31;
    const int w    = tid >> 5;
    const int m0   = blockIdx.x * 128;
    const int n0   = blockIdx.y * 64;

    // --- TDM descriptor group 1 (loop-invariant) ---
    // data_size=2B, tile 32 x 128, tensor dims huge (no clipping), stride=K
    u32x8 g1;
    g1[0] = 0x00010000u;                    // wg_mask=0, data_size=1 (2B)
    g1[1] = 0xffff0000u;                    // tensor_dim0 lo16 = 0xffff
    g1[2] = 0x7fffu | 0xffff0000u;          // dim0 hi=0x7fff, dim1 lo=0xffff
    g1[3] = 0x7fffu | (32u << 16);          // dim1 hi, tile_dim0 = 32
    g1[4] = 128u;                           // tile_dim1 = 128, tile_dim2 = 0
    g1[5] = (unsigned)K;                    // tensor_dim0_stride lo32
    g1[6] = 0u;                             // stride hi16, dim1_stride lo16
    g1[7] = 0u;
    const unsigned ldsA = lds_offset(&As[0]);

    // --- per-thread B staging pointers (8 bf16 each, x2) ---
    // lin = j*128 + tid : bk = lin>>3 (k row 0..31), nq = lin&7 (n octet)
    const __bf16* pb0 = B + (size_t)(tid >> 3) * N + n0 + (tid & 7) * 8;
    const __bf16* pb1 = pb0 + (size_t)16 * N;   // +128 lanes = +16 k rows
    const int bk0 = tid >> 3, nq0 = (tid & 7) * 8;

    v8f acc[2][4] = {};

    for (int k0 = 0; k0 < K; k0 += 32) {
        __syncthreads();
        if (w == 0) {
            unsigned long long ga =
                (unsigned long long)(const void*)(A + (size_t)m0 * K + k0);
            u32x4 g0;
            g0[0] = 1u;                                   // count=1, user D#
            g0[1] = ldsA;                                 // LDS dest
            g0[2] = (unsigned)ga;                         // global addr lo
            g0[3] = ((unsigned)(ga >> 32) & 0x01ffffffu)  // global addr hi
                    | 0x80000000u;                        // type = 2 (image)
            asm volatile("tensor_load_to_lds %0, %1"
                         :: "s"(g0), "s"(g1) : "memory");
        }
        {   // B tile: 2 x uint4 load, transposed store
            union { uint4 u; __bf16 h[8]; } bv;
            bv.u = *(const uint4*)pb0;
#pragma unroll
            for (int e = 0; e < 8; ++e) Bt[(nq0 + e) * 32 + bk0] = bv.h[e];
            bv.u = *(const uint4*)pb1;
#pragma unroll
            for (int e = 0; e < 8; ++e) Bt[(nq0 + e) * 32 + bk0 + 16] = bv.h[e];
            pb0 += (size_t)32 * N;
            pb1 += (size_t)32 * N;
        }
        if (w == 0) __builtin_amdgcn_s_wait_tensorcnt(0);
        __syncthreads();

        v16bf bfrag[4];
#pragma unroll
        for (int nt = 0; nt < 4; ++nt)
            bfrag[nt] = load_bt_frag(Bt + nt * 16 * 32, 32);
#pragma unroll
        for (int sub = 0; sub < 2; ++sub) {
            v16bf a = load_a_frag(As + (w * 32 + sub * 16) * 32, 32);
#pragma unroll
            for (int nt = 0; nt < 4; ++nt)
                acc[sub][nt] = wmma_bf16(a, bfrag[nt], acc[sub][nt]);
        }
    }

    const int col = lane & 15;
    const int hi8 = (lane >> 4) << 3;

    if (mode == 0) {
#pragma unroll
        for (int sub = 0; sub < 2; ++sub)
#pragma unroll
            for (int r = 0; r < 8; ++r) {
                int row = m0 + w * 32 + sub * 16 + r + hi8;
#pragma unroll
                for (int nt = 0; nt < 4; ++nt)
                    C[(size_t)row * N + n0 + nt * 16 + col] = acc[sub][nt][r];
            }
    } else {
        const int region = n0 >> 10;          // E = 1024: q / k / v section
        const int h      = (n0 & 1023) >> 6;  // head (hd = 64)
#pragma unroll
        for (int sub = 0; sub < 2; ++sub)
#pragma unroll
            for (int r = 0; r < 8; ++r) {
                int m = m0 + w * 32 + sub * 16 + r + hi8;
                int b = m >> 11;              // T = 2048
                int t = m & 2047;
                size_t base = (((size_t)b * 16 + h) * 2048 + t) * 64;
#pragma unroll
                for (int nt = 0; nt < 4; ++nt) {
                    int   d = nt * 16 + col;
                    float v = acc[sub][nt][r];
                    if (region == 2) {
                        Vb[base + d] = (__bf16)v;
                    } else {
                        float cc = cosp[t * 64 + d];
                        float ss = sinp[t * 64 + d];
                        float partner = acc[sub][nt ^ 2][r]; // column d ^ 32
                        float rh  = (d < 32) ? -partner : partner;
                        float val = v * cc + rh * ss;
                        if (region == 0)
                            Qb[base + d] = (__bf16)(val * 0.125f); // 1/sqrt(64)
                        else
                            Kb[base + d] = (__bf16)val;
                    }
                }
            }
    }
}

// ---------------------------------------------------------------------------
// Causal attention, flash-style.  grid = (B*H, T/64), 128 threads = 4 waves.
// K tile (contiguous 4KB) streamed with global_load_async_to_lds_b128
// (ASYNCcnt); V staged with uint4 loads + transpose; online softmax with
// 16-lane butterfly shuffles; P relayout via per-wave LDS.
// ---------------------------------------------------------------------------
__global__ __launch_bounds__(128) void attn_kernel(
    const __bf16* __restrict__ Qb, const __bf16* __restrict__ Kb,
    const __bf16* __restrict__ Vb, __bf16* __restrict__ Ob)
{
    __shared__ __bf16 Ks[32 * 64];     // [key][hd]
    __shared__ __bf16 Vt[64 * 32];     // [hd][key]
    __shared__ __bf16 Pb[4][16 * 32];  // per-wave P tile, [q][key]

    const int tid  = threadIdx.x;
    const int lane = tid & 31;
    const int w    = tid >> 5;
    const int bh   = blockIdx.x;       // b*16 + h
    const int q0   = blockIdx.y * 64;
    const int col  = lane & 15;
    const int hi8  = (lane >> 4) << 3;

    const __bf16* qbase = Qb + ((size_t)bh * 2048 + q0 + w * 16) * 64;
    v16bf qa0 = load_a_frag(qbase, 64);        // hd 0..31
    v16bf qa1 = load_a_frag(qbase + 32, 64);   // hd 32..63

    v8f acc[4] = {};
    float mrow[8], lrow[8];
#pragma unroll
    for (int r = 0; r < 8; ++r) { mrow[r] = -1e30f; lrow[r] = 0.f; }

    const unsigned ldsK = lds_offset(&Ks[0]);
    const int vkey = tid >> 3, vdq = (tid & 7) * 8;   // V staging coords

    for (int kb = 0; kb < q0 + 64; kb += 32) {
        __syncthreads();
        const __bf16* kg = Kb + ((size_t)bh * 2048 + kb) * 64;
        const __bf16* vg = Vb + ((size_t)bh * 2048 + kb) * 64;

        // K tile: 2 async 16B copies per thread (256 x 16B = 4KB)
        {
            unsigned long long ga = (unsigned long long)(const void*)(kg + tid * 8);
            unsigned la = ldsK + tid * 16;
            asm volatile("global_load_async_to_lds_b128 %0, %1, off"
                         :: "v"(la), "v"(ga) : "memory");
            ga += 128ull * 16ull;
            la += 128u * 16u;
            asm volatile("global_load_async_to_lds_b128 %0, %1, off"
                         :: "v"(la), "v"(ga) : "memory");
        }
        // V tile: uint4 loads + transposed stores (Vt[hd][key])
        {
            union { uint4 u; __bf16 h[8]; } vv;
            vv.u = *(const uint4*)(vg + vkey * 64 + vdq);
#pragma unroll
            for (int e = 0; e < 8; ++e) Vt[(vdq + e) * 32 + vkey] = vv.h[e];
            vv.u = *(const uint4*)(vg + (vkey + 16) * 64 + vdq);
#pragma unroll
            for (int e = 0; e < 8; ++e) Vt[(vdq + e) * 32 + vkey + 16] = vv.h[e];
        }
        asm volatile("s_wait_asynccnt 0x0" ::: "memory");
        __syncthreads();

        // S tile 16x32: two 16x16 C tiles, hd contraction in two chunks
        v8f s[2];
#pragma unroll
        for (int nt = 0; nt < 2; ++nt) {
            v8f z = {};
            z = wmma_bf16(qa0, load_bt_frag(Ks + nt * 16 * 64,      64), z);
            z = wmma_bf16(qa1, load_bt_frag(Ks + nt * 16 * 64 + 32, 64), z);
            s[nt] = z;
        }

        // causal mask + online softmax (butterfly within 16-lane half)
#pragma unroll
        for (int r = 0; r < 8; ++r) {
            int row = q0 + w * 16 + r + hi8;
            float s0 = s[0][r], s1 = s[1][r];
            if (kb + col      > row) s0 = -1e30f;
            if (kb + 16 + col > row) s1 = -1e30f;
            float mloc = fmaxf(s0, s1);
#pragma unroll
            for (int msk = 1; msk < 16; msk <<= 1)
                mloc = fmaxf(mloc, __shfl_xor(mloc, msk, 32));
            float mnew  = fmaxf(mrow[r], mloc);
            float alpha = __expf(mrow[r] - mnew);
            float p0 = __expf(s0 - mnew);
            float p1 = __expf(s1 - mnew);
            float rs = p0 + p1;
#pragma unroll
            for (int msk = 1; msk < 16; msk <<= 1)
                rs += __shfl_xor(rs, msk, 32);
            lrow[r] = lrow[r] * alpha + rs;
            mrow[r] = mnew;
#pragma unroll
            for (int nt = 0; nt < 4; ++nt) acc[nt][r] *= alpha;
            Pb[w][(r + hi8) * 32 + col]      = (__bf16)p0;
            Pb[w][(r + hi8) * 32 + 16 + col] = (__bf16)p1;
        }

        // O += P(16x32) x V(32x64); same-wave LDS ops are in-order
        v16bf pa = load_a_frag(&Pb[w][0], 32);
#pragma unroll
        for (int nt = 0; nt < 4; ++nt)
            acc[nt] = wmma_bf16(pa, load_bt_frag(Vt + nt * 16 * 32, 32), acc[nt]);
    }

    // normalize, write O as (B, T, E) bf16 for the out-projection GEMM
    const int b = bh >> 4, h = bh & 15;
#pragma unroll
    for (int r = 0; r < 8; ++r) {
        int   t   = q0 + w * 16 + r + hi8;
        float inv = 1.0f / lrow[r];
        size_t base = ((size_t)b * 2048 + t) * 1024 + h * 64;
#pragma unroll
        for (int nt = 0; nt < 4; ++nt)
            Ob[base + nt * 16 + col] = (__bf16)(acc[nt][r] * inv);
    }
}

// ---------------------------------------------------------------------------
// Launch: convert -> QKV GEMM (+RoPE epilogue) -> attention -> out-proj GEMM
// ---------------------------------------------------------------------------
extern "C" void kernel_launch(void* const* d_in, const int* in_sizes, int n_in,
                              void* d_out, int out_size, void* d_ws, size_t ws_size,
                              hipStream_t stream)
{
    const float* x     = (const float*)d_in[0];   // (2, 2048, 1024)
    const float* cosp  = (const float*)d_in[1];   // (2048, 64)
    const float* sinp  = (const float*)d_in[2];   // (2048, 64)
    const float* w_qkv = (const float*)d_in[3];   // (1024, 3072)
    const float* w_out = (const float*)d_in[4];   // (1024, 1024)
    float* out = (float*)d_out;

    char* ws = (char*)d_ws;
    __bf16* Xb  = (__bf16*)(ws);              //  8 MB : x bf16
    __bf16* Wqb = (__bf16*)(ws + 8388608);    //  6 MB : w_qkv bf16
    __bf16* Wob = (__bf16*)(ws + 14680064);   //  2 MB : w_out bf16
    __bf16* Qb  = (__bf16*)(ws + 16777216);   //  8 MB : Q (B,H,T,hd)
    __bf16* Kb  = (__bf16*)(ws + 25165824);   //  8 MB : K
    __bf16* Vb  = (__bf16*)(ws + 33554432);   //  8 MB : V
    __bf16* Ob  = (__bf16*)(ws + 41943040);   //  8 MB : attn out (B,T,E)

    f32_to_bf16_kernel<<<(2 * 2048 * 1024 / 8 + 255) / 256, 256, 0, stream>>>(
        x, Xb, 2 * 2048 * 1024 / 8);
    f32_to_bf16_kernel<<<(1024 * 3072 / 8 + 255) / 256, 256, 0, stream>>>(
        w_qkv, Wqb, 1024 * 3072 / 8);
    f32_to_bf16_kernel<<<(1024 * 1024 / 8 + 255) / 256, 256, 0, stream>>>(
        w_out, Wob, 1024 * 1024 / 8);

    // QKV projection with fused RoPE epilogue: M=4096, N=3072, K=1024
    gemm_bf16_kernel<<<dim3(32, 48), 128, 0, stream>>>(
        Xb, Wqb, 4096, 3072, 1024, 1, nullptr, Qb, Kb, Vb, cosp, sinp);

    // attention: (B*H = 32, T/64 = 32)
    attn_kernel<<<dim3(32, 32), 128, 0, stream>>>(Qb, Kb, Vb, Ob);

    // out projection: M=4096, N=1024, K=1024, f32 result
    gemm_bf16_kernel<<<dim3(32, 16), 128, 0, stream>>>(
        Ob, Wob, 4096, 1024, 1024, 0, out, nullptr, nullptr, nullptr, nullptr, nullptr);
}
